// FlashAttention_90735479095802
// MI455X (gfx1250) — matrix operand order repaired and checked by reading, not compiled, not run on hardware
//
#include <hip/hip_runtime.h>

typedef __attribute__((ext_vector_type(16))) _Float16 v16h;
typedef __attribute__((ext_vector_type(8)))  float    v8f;

#define E_DIM 1024
#define MTOT  16384   // B*S = 4*4096

union AFrag { v16h v; uint4 u[2]; };

static __device__ __forceinline__ v8f wmma_f16(v16h a, v16h b, v8f c) {
  // D = A(16x32) * B(32x16) + C, f32 accumulate
  return __builtin_amdgcn_wmma_f32_16x16x32_f16(false, a, false, b, (short)0, c,
                                                false, false);
}

// Low 32 bits of a generic pointer to __shared__ = LDS byte offset (flat
// aperture truncation, ISA 10.2).
static __device__ __forceinline__ uint32_t lds_addr(const void* p) {
  return (uint32_t)(uintptr_t)p;
}

// CDNA5 async tensor-path staging: global -> LDS without VGPR round trip.
// VDST = LDS byte address, VADDR = 64-bit global address. Tracked by ASYNCcnt.
static __device__ __forceinline__ void async_ld_b128(uint32_t lds, const void* g) {
  asm volatile("global_load_async_to_lds_b128 %0, %1, off"
               :: "v"(lds), "v"(g) : "memory");
}

#define S_WAIT_ASYNCCNT(n) asm volatile("s_wait_asynccnt %0" :: "n"(n) : "memory")

// ---------------------------------------------------------------------------
// f32 -> f16 conversion (grid-stride)
// ---------------------------------------------------------------------------
__global__ void cvt_f32_to_f16(const float* __restrict__ in,
                               _Float16* __restrict__ out, int n) {
  int i = blockIdx.x * blockDim.x + threadIdx.x;
  int stride = gridDim.x * blockDim.x;
  for (; i < n; i += stride) out[i] = (_Float16)in[i];
}

// wt[k*E + n] = (f16) w[n*E + k]   (coalesced reads of w)
__global__ void cvt_transpose_w(const float* __restrict__ w,
                                _Float16* __restrict__ wt) {
  int idx = blockIdx.x * blockDim.x + threadIdx.x; // idx = n*E + k
  int nrow = idx >> 10;
  int k    = idx & 1023;
  wt[k * E_DIM + nrow] = (_Float16)w[idx];
}

// ---------------------------------------------------------------------------
// Tiled GEMM: Y[M,1024] = X[M,1024] @ Wt[1024,1024]   (Wt is W^T, K-major rows)
// 256x128 tile per workgroup, 8 waves; each wave owns 32 rows x 128 cols
// (two A-frags amortize every B-frag load over 2 WMMAs).
// Double-buffered LDS filled by async-to-LDS loads (software pipeline).
// ---------------------------------------------------------------------------
static __device__ __forceinline__ void stage_gemm_tile(
    const _Float16* __restrict__ X, const _Float16* __restrict__ Wt,
    int mBase, int nBase, int kc, int tid,
    _Float16 (*__restrict__ Xs)[40], _Float16 (*__restrict__ Ws)[136]) {
  // X tile: 256 rows x 32 K-halves = 1024 b128 chunks, 4 per thread
  for (int i = 0; i < 4; ++i) {
    int c = tid + i * 256;
    int r = c >> 2, c8 = (c & 3) * 8;
    async_ld_b128(lds_addr(&Xs[r][c8]),
                  X + (size_t)(mBase + r) * E_DIM + kc * 32 + c8);
  }
  // W tile: 32 K-rows x 128 N-halves = 512 b128 chunks, 2 per thread
  for (int i = 0; i < 2; ++i) {
    int c = tid + i * 256;
    int r = c >> 4, c8 = (c & 15) * 8;
    async_ld_b128(lds_addr(&Ws[r][c8]),
                  Wt + (size_t)(kc * 32 + r) * E_DIM + nBase + c8);
  }
}

template <bool OUT_F32>
__global__ __launch_bounds__(256) void gemm256(const _Float16* __restrict__ X,
                                               const _Float16* __restrict__ Wt,
                                               void* __restrict__ Yv) {
  __shared__ __align__(16) _Float16 Xs[2][256][40];   // 80B row stride (16B mult)
  __shared__ __align__(16) _Float16 Ws[2][32][136];   // 272B row stride

  const int tid  = threadIdx.x;
  const int lane = tid & 31;
  const int wave = tid >> 5;
  const int mBase = blockIdx.y * 256;
  const int nBase = blockIdx.x * 128;

  v8f acc0[8] = {}, acc1[8] = {};

  const int arow = wave * 32 + (lane & 15);
  const int kb   = (lane < 16) ? 0 : 8;           // A-frag K-chunk select

  stage_gemm_tile(X, Wt, mBase, nBase, 0, tid, Xs[0], Ws[0]);

  for (int kc = 0; kc < E_DIM / 32; ++kc) {
    const int cur = kc & 1;
    __syncthreads();                       // prev compute done: buf cur^1 is free
    if (kc + 1 < E_DIM / 32) {
      stage_gemm_tile(X, Wt, mBase, nBase, kc + 1, tid, Xs[cur ^ 1], Ws[cur ^ 1]);
      S_WAIT_ASYNCCNT(6);                  // tile kc's 6 loads retired
    } else {
      S_WAIT_ASYNCCNT(0);
    }
    __syncthreads();                       // tile kc visible from all waves

    AFrag a0, a1;
    a0.u[0] = *(const uint4*)&Xs[cur][arow][kb];
    a0.u[1] = *(const uint4*)&Xs[cur][arow][kb + 16];
    a1.u[0] = *(const uint4*)&Xs[cur][arow + 16][kb];
    a1.u[1] = *(const uint4*)&Xs[cur][arow + 16][kb + 16];
    for (int nt = 0; nt < 8; ++nt) {
      AFrag b;                                    // B lane = N-col, elems = K (table order)
      for (int s = 0; s < 16; ++s)
        b.v[s] = Ws[cur][kb + (s & 7) + (s >> 3) * 16][nt * 16 + (lane & 15)];
      acc0[nt] = wmma_f16(a0.v, b.v, acc0[nt]);
      acc1[nt] = wmma_f16(a1.v, b.v, acc1[nt]);
    }
  }

  const int rowLocal = wave * 32 + ((lane < 16) ? 0 : 8);
  const int colLane  = lane & 15;
  if (OUT_F32) {
    float* Y = (float*)Yv;
    for (int nt = 0; nt < 8; ++nt)
      for (int j = 0; j < 8; ++j) {
        size_t col = (size_t)nBase + nt * 16 + colLane;
        Y[(size_t)(mBase + rowLocal + j) * E_DIM + col]      = acc0[nt][j];
        Y[(size_t)(mBase + rowLocal + 16 + j) * E_DIM + col] = acc1[nt][j];
      }
  } else {
    _Float16* Y = (_Float16*)Yv;
    for (int nt = 0; nt < 8; ++nt)
      for (int j = 0; j < 8; ++j) {
        size_t col = (size_t)nBase + nt * 16 + colLane;
        Y[(size_t)(mBase + rowLocal + j) * E_DIM + col]      = (_Float16)acc0[nt][j];
        Y[(size_t)(mBase + rowLocal + 16 + j) * E_DIM + col] = (_Float16)acc1[nt][j];
      }
  }
}

// ---------------------------------------------------------------------------
// Block-local attention: one workgroup per (b, h, n) 128x128 block.
// S = (Q K^T) * 0.125 ; P = softmax_row(S) ; O = P V  — all via WMMA f16.
// V staged via async-to-LDS; K staged transposed (b128 loads + b16 stores).
// ---------------------------------------------------------------------------
__global__ __launch_bounds__(256) void block_attn(const _Float16* __restrict__ Q,
                                                  const _Float16* __restrict__ K,
                                                  const _Float16* __restrict__ V,
                                                  _Float16* __restrict__ O) {
  __shared__ __align__(16) _Float16 Kt[64][136];   // K transposed: [d][s]
  __shared__ __align__(16) _Float16 Vs[128][72];   // V: [s][d], 144B stride
  __shared__ __align__(16) _Float16 Ps[128][136];  // softmax probs: [t][s]

  const int tid  = threadIdx.x;
  const int lane = tid & 31;
  const int wave = tid >> 5;
  const int blk  = blockIdx.x;           // b*512 + h*32 + n
  const int nblk = blk & 31;
  const int h    = (blk >> 5) & 15;
  const int bat  = blk >> 9;
  const size_t rowBase = (size_t)bat * 4096 + (size_t)nblk * 128;
  const int colBase = h * 64;

  // V: 128x64 halves = 1024 b128 chunks, async straight into LDS
  for (int i = 0; i < 4; ++i) {
    int c = tid + i * 256;
    int s = c >> 3, d8 = (c & 7) * 8;
    async_ld_b128(lds_addr(&Vs[s][d8]),
                  V + (rowBase + s) * E_DIM + colBase + d8);
  }
  // K: vector global loads, transposed scalar LDS stores (overlaps with async V)
  for (int i = 0; i < 4; ++i) {
    int c = tid + i * 256;
    int s = c >> 3, d8 = (c & 7) * 8;
    uint4 kv = *(const uint4*)(K + (rowBase + s) * E_DIM + colBase + d8);
    const _Float16* kh = (const _Float16*)&kv;
    for (int j = 0; j < 8; ++j) Kt[d8 + j][s] = kh[j];
  }
  S_WAIT_ASYNCCNT(0);
  __syncthreads();

  const int mrow = wave * 16 + (lane & 15);
  const int kb   = (lane < 16) ? 0 : 8;

  // Q A-fragments straight from global (aligned 16B chunks)
  AFrag aq[2];
  for (int kc = 0; kc < 2; ++kc) {
    const _Float16* qrow = Q + (rowBase + mrow) * E_DIM + colBase + kc * 32;
    aq[kc].u[0] = *(const uint4*)(qrow + kb);
    aq[kc].u[1] = *(const uint4*)(qrow + kb + 16);
  }

  // S = Q K^T : 16x128 strip per wave, K-dim 64 = 2 WMMA chunks
  v8f st[8] = {};
  for (int kc = 0; kc < 2; ++kc)
    for (int t = 0; t < 8; ++t) {
      AFrag bk;                                  // B lane = key col, elems = feature (table order)
      for (int s = 0; s < 16; ++s)
        bk.v[s] = Kt[kc * 32 + kb + (s & 7) + (s >> 3) * 16][t * 16 + (lane & 15)];
      st[t] = wmma_f16(aq[kc].v, bk.v, st[t]);
    }

  // row softmax: VGPR j holds rows j (lanes 0-15) / j+8 (lanes 16-31);
  // 16 lanes of a half-group hold the 16 columns of one 16-col tile.
  const float scale = 0.125f;  // 1/sqrt(64)
  for (int j = 0; j < 8; ++j) {
    float m = -1e30f;
    for (int t = 0; t < 8; ++t) { st[t][j] *= scale; m = fmaxf(m, st[t][j]); }
    for (int msk = 1; msk < 16; msk <<= 1) m = fmaxf(m, __shfl_xor(m, msk, 32));
    float sum = 0.0f;
    for (int t = 0; t < 8; ++t) {
      float p = __expf(st[t][j] - m);
      st[t][j] = p;
      sum += p;
    }
    for (int msk = 1; msk < 16; msk <<= 1) sum += __shfl_xor(sum, msk, 32);
    float r = 1.0f / sum;
    int prow = wave * 16 + j + ((lane < 16) ? 0 : 8);
    for (int t = 0; t < 8; ++t)
      Ps[prow][t * 16 + (lane & 15)] = (_Float16)(st[t][j] * r);
  }
  // Ps written/read by the same wave only -> same-wave LDS ordering suffices.

  // O = P V : 16x64 strip per wave, K-dim 128 = 4 WMMA chunks
  v8f od[4] = {};
  for (int kt = 0; kt < 4; ++kt) {
    AFrag ap;
    ap.u[0] = *(const uint4*)&Ps[mrow][kt * 32 + kb];
    ap.u[1] = *(const uint4*)&Ps[mrow][kt * 32 + kb + 16];
    for (int dt = 0; dt < 4; ++dt) {
      AFrag bv;
      for (int s = 0; s < 16; ++s)
        bv.v[s] = Vs[kt * 32 + kb + (s & 7) + (s >> 3) * 16][dt * 16 + (lane & 15)];
      od[dt] = wmma_f16(ap.v, bv.v, od[dt]);
    }
  }

  const int orow = wave * 16 + ((lane < 16) ? 0 : 8);
  for (int dt = 0; dt < 4; ++dt)
    for (int j = 0; j < 8; ++j)
      O[(rowBase + orow + j) * E_DIM + colBase + dt * 16 + (lane & 15)] =
          (_Float16)od[dt][j];
}

// ---------------------------------------------------------------------------
extern "C" void kernel_launch(void* const* d_in, const int* in_sizes, int n_in,
                              void* d_out, int out_size, void* d_ws, size_t ws_size,
                              hipStream_t stream) {
  const float* x  = (const float*)d_in[0];
  const float* Wq = (const float*)d_in[1];
  const float* Wk = (const float*)d_in[2];
  const float* Wv = (const float*)d_in[3];
  const float* Wo = (const float*)d_in[4];

  const size_t nX = (size_t)MTOT * E_DIM;   // 16,777,216 elements
  const size_t nW = (size_t)E_DIM * E_DIM;  // 1,048,576 elements

  _Float16* xh  = (_Float16*)d_ws;
  _Float16* qb  = xh  + nX;
  _Float16* kb  = qb  + nX;
  _Float16* vb  = kb  + nX;
  _Float16* wtq = vb  + nX;
  _Float16* wtk = wtq + nW;
  _Float16* wtv = wtk + nW;
  _Float16* wto = wtv + nW;
  _Float16* ao  = xh;   // xh is dead after the QKV projections; reuse as attn out

  cvt_f32_to_f16<<<4096, 256, 0, stream>>>(x, xh, (int)nX);
  cvt_transpose_w<<<(int)(nW / 256), 256, 0, stream>>>(Wq, wtq);
  cvt_transpose_w<<<(int)(nW / 256), 256, 0, stream>>>(Wk, wtk);
  cvt_transpose_w<<<(int)(nW / 256), 256, 0, stream>>>(Wv, wtv);
  cvt_transpose_w<<<(int)(nW / 256), 256, 0, stream>>>(Wo, wto);

  dim3 g(E_DIM / 128, MTOT / 256);  // (8, 64)
  gemm256<false><<<g, 256, 0, stream>>>(xh, wtq, (void*)qb);
  gemm256<false><<<g, 256, 0, stream>>>(xh, wtk, (void*)kb);
  gemm256<false><<<g, 256, 0, stream>>>(xh, wtv, (void*)vb);

  block_attn<<<2048, 256, 0, stream>>>(qb, kb, vb, ao);

  gemm256<true><<<g, 256, 0, stream>>>(ao, wto, d_out);
}
